// ArcAwareAttention_19928648253536
// MI455X (gfx1250) — matrix operand order repaired and checked by reading, not compiled, not run on hardware
//
#include <hip/hip_runtime.h>

typedef __attribute__((ext_vector_type(16))) __bf16 v16bf;
typedef __attribute__((ext_vector_type(8)))  __bf16 v8bf;
typedef __attribute__((ext_vector_type(4)))  __bf16 v4bf;
typedef __attribute__((ext_vector_type(8)))  float  v8f;
typedef __attribute__((ext_vector_type(4)))  float  v4f;

#define BATCH 2
#define SEQ   4096
#define CH    256
#define NH    4
#define HD    64
#define QKVLD 768          // 3*CH row stride of fused qkv
#define SCALE 0.125f       // d^-0.5, d=64
#define PSTR  0.1f         // ARC_PRIOR_STRENGTH

// ---------- WMMA helpers (wave32, 16x16x32 bf16 -> f32) ----------

__device__ __forceinline__ v8f wmma_bf16(v16bf a, v16bf b, v8f c) {
  // (neg_a, A, neg_b, B, c_mod, C, reuse_a, reuse_b)
  return __builtin_amdgcn_wmma_f32_16x16x32_bf16(false, a, false, b, (short)0, c,
                                                 false, false);
}

__device__ __forceinline__ v16bf cat8(v8bf lo, v8bf hi) {
  return __builtin_shufflevector(lo, hi, 0, 1, 2, 3, 4, 5, 6, 7,
                                 8, 9, 10, 11, 12, 13, 14, 15);
}

// A-fragment (16x32 bf16, row-major source): lane holds row m = lane&15,
// K = 16*(lane>=16)+e  -> one 32-byte load.
__device__ __forceinline__ v16bf load_a_frag(const __bf16* rowptr, int lane) {
  return *(const v16bf*)(rowptr + ((lane >> 4) << 4));
}
// B-fragment (32x16 bf16, K-contiguous source): lane holds col n = lane&15,
// K = 16*(lane>=16)+e  -> one 32-byte load.
__device__ __forceinline__ v16bf load_b_frag(const __bf16* colptr, int lane) {
  return *(const v16bf*)(colptr + ((lane >> 4) << 4));
}
// C/D (16x16 f32): row = r + 8*(lane>=16), col = lane&15

// ---------- conversion / transpose prep kernels ----------

__global__ void k_xcvt(const float* __restrict__ x, __bf16* __restrict__ xb) {
  int i = blockIdx.x * blockDim.x + threadIdx.x;       // over (B*SEQ*CH)/4
  v4f v = ((const v4f*)x)[i];
  v4bf o;
#pragma unroll
  for (int e = 0; e < 4; ++e) o[e] = (__bf16)v[e];
  ((v4bf*)xb)[i] = o;
}

// W[k][col] f32 -> WT[col][k] bf16   (ldw = row length of W)
__global__ void k_wt(const float* __restrict__ W, __bf16* __restrict__ WT, int ldw) {
  int i = blockIdx.x * blockDim.x + threadIdx.x;       // [0, ldw*CH)
  int col = i >> 8, k = i & 255;                       // K = CH = 256
  WT[i] = (__bf16)W[(size_t)k * ldw + col];
}

// qkv v-slice -> vT[((b*NH+h)*HD + ch)][tok] bf16 (token-contiguous)
__global__ void k_vt(const __bf16* __restrict__ qkv, __bf16* __restrict__ vT) {
  int i = blockIdx.x * blockDim.x + threadIdx.x;       // [0, B*NH*HD*SEQ)
  int tok = i & (SEQ - 1);
  int rest = i >> 12;
  int ch = rest & (HD - 1);
  int h = (rest >> 6) & (NH - 1);
  int b = rest >> 8;
  vT[i] = qkv[(size_t)(b * SEQ + tok) * QKVLD + 2 * CH + h * HD + ch];
}

// ---------- K1: channel mean  x[B,N,C] -> xm[B*N] ----------
__global__ void k_mean(const float* __restrict__ x, float* __restrict__ xm) {
  int i = blockIdx.x * blockDim.x + threadIdx.x;       // [0, B*SEQ)
  const float* p = x + (size_t)i * CH;
  float s = 0.f;
  for (int c = 0; c < CH; ++c) s += p[c];
  xm[i] = s * (1.0f / CH);
}

// ---------- K2: arc prior = sigmoid(conv3x3 + conv5x5), SAME zero pad ----------
__global__ void k_prior(const float* __restrict__ xm, const float* __restrict__ ak,
                        const float* __restrict__ ck, float* __restrict__ prior) {
  int i = blockIdx.x * blockDim.x + threadIdx.x;       // [0, B*SEQ)
  int b = i / SEQ, idx = i % SEQ;
  int yy = idx >> 6, xx = idx & 63;
  const float* base = xm + b * SEQ;
  float s = 0.f;
  for (int dy = -1; dy <= 1; ++dy)
    for (int dx = -1; dx <= 1; ++dx) {
      int y = yy + dy, x2 = xx + dx;
      if ((unsigned)y < 64u && (unsigned)x2 < 64u)
        s += ak[(dy + 1) * 3 + (dx + 1)] * base[y * 64 + x2];
    }
  for (int dy = -2; dy <= 2; ++dy)
    for (int dx = -2; dx <= 2; ++dx) {
      int y = yy + dy, x2 = xx + dx;
      if ((unsigned)y < 64u && (unsigned)x2 < 64u)
        s += ck[(dy + 2) * 5 + (dx + 2)] * base[y * 64 + x2];
    }
  prior[i] = 1.f / (1.f + __expf(-s));
}

// ---------- K3: qkv = x @ Wqkv + b  ->  bf16 [B*N, 768] ----------
__global__ void k_qkv(const __bf16* __restrict__ xb, const __bf16* __restrict__ WT,
                      const float* __restrict__ bias, __bf16* __restrict__ qkv) {
  int lane = threadIdx.x & 31;
  int wid = blockIdx.x * 8 + (threadIdx.x >> 5);   // one wave = one 16x16 tile
  int nt = wid % 48, rt = wid / 48;                // 48 col tiles, 512 row tiles
  int m = lane & 15;
  int row = rt * 16 + m;
  int col = nt * 16 + m;
  v8f acc = {};
  for (int kt = 0; kt < 8; ++kt) {                 // K = 256 = 8 * 32
    v16bf a = load_a_frag(xb + (size_t)row * CH + kt * 32, lane);
    v16bf b = load_b_frag(WT + (size_t)col * CH + kt * 32, lane);
    acc = wmma_bf16(a, b, acc);
  }
  float bv = bias[col];
#pragma unroll
  for (int r = 0; r < 8; ++r) {
    int orow = rt * 16 + r + ((lane >> 4) << 3);
    qkv[(size_t)orow * QKVLD + col] = (__bf16)(acc[r] + bv);
  }
}

// ---------- K4: pass-1 flash stats: row max + sum(exp) per (b,h,row) ----------
__global__ void k_stats(const __bf16* __restrict__ qkv, const float* __restrict__ prior,
                        float* __restrict__ rm, float* __restrict__ rl) {
  int lane = threadIdx.x & 31;
  int wid = blockIdx.x * 8 + (threadIdx.x >> 5);   // [0, 2048)
  int b = wid >> 10, h = (wid >> 8) & 3, rt = wid & 255;
  int m = lane & 15;
  v16bf qa[2];
#pragma unroll
  for (int ks = 0; ks < 2; ++ks)
    qa[ks] = load_a_frag(qkv + (size_t)(b * SEQ + rt * 16 + m) * QKVLD + h * HD +
                         ks * 32, lane);
  float pr[8], mx[8], ls[8];
#pragma unroll
  for (int r = 0; r < 8; ++r) {
    pr[r] = prior[b * SEQ + rt * 16 + r + ((lane >> 4) << 3)] * PSTR;
    mx[r] = -3.0e38f; ls[r] = 0.f;
  }
  for (int ct = 0; ct < 256; ++ct) {
    v8f acc = {};
    int tok = ct * 16 + m;
    const __bf16* kptr = qkv + (size_t)(b * SEQ + tok) * QKVLD + CH + h * HD;
    __builtin_prefetch((const void*)(kptr + 16 * QKVLD), 0, 1);
#pragma unroll
    for (int ks = 0; ks < 2; ++ks)
      acc = wmma_bf16(qa[ks], load_b_frag(kptr + ks * 32, lane), acc);
    float pc = prior[b * SEQ + ct * 16 + m];
#pragma unroll
    for (int r = 0; r < 8; ++r) {
      float sv = acc[r] * SCALE + pr[r] * pc;
      float tm = sv;
      for (int msk = 8; msk >= 1; msk >>= 1) tm = fmaxf(tm, __shfl_xor(tm, msk, 32));
      float mn = fmaxf(mx[r], tm);
      float ev = __expf(sv - mn);
      float rs = ev;
      for (int msk = 8; msk >= 1; msk >>= 1) rs += __shfl_xor(rs, msk, 32);
      ls[r] = ls[r] * __expf(mx[r] - mn) + rs;
      mx[r] = mn;
    }
  }
  if (m == 0) {
#pragma unroll
    for (int r = 0; r < 8; ++r) {
      int row = rt * 16 + r + ((lane >> 4) << 3);
      size_t o = (size_t)(b * NH + h) * SEQ + row;
      rm[o] = mx[r];
      rl[o] = ls[r];
    }
  }
}

// ---------- K5: pass-2 P@V with P staged through LDS -> head_out bf16 ----------
__global__ void k_av(const __bf16* __restrict__ qkv, const __bf16* __restrict__ vT,
                     const float* __restrict__ prior, const float* __restrict__ rm,
                     const float* __restrict__ rl, __bf16* __restrict__ houtb) {
  __shared__ __bf16 pb[8][16][40];                 // per-wave 16x32 P tile, 16B-aligned rows
  int lane = threadIdx.x & 31;
  int w = threadIdx.x >> 5;
  int wid = blockIdx.x * 8 + w;                    // [0, 2048)
  int b = wid >> 10, h = (wid >> 8) & 3, rt = wid & 255;
  int m = lane & 15;
  v16bf qa[2];
#pragma unroll
  for (int ks = 0; ks < 2; ++ks)
    qa[ks] = load_a_frag(qkv + (size_t)(b * SEQ + rt * 16 + m) * QKVLD + h * HD +
                         ks * 32, lane);
  float pr[8], pm[8], pl[8];
#pragma unroll
  for (int r = 0; r < 8; ++r) {
    int row = rt * 16 + r + ((lane >> 4) << 3);
    pr[r] = prior[b * SEQ + row] * PSTR;
    size_t o = (size_t)(b * NH + h) * SEQ + row;
    pm[r] = rm[o];
    pl[r] = 1.f / rl[o];
  }
  v8f oacc[4] = {};
  for (int c2 = 0; c2 < 128; ++c2) {               // 32 key tokens per step
#pragma unroll
    for (int t = 0; t < 2; ++t) {
      int ct = c2 * 2 + t;
      v8f acc = {};
      int tok = ct * 16 + m;
      const __bf16* kptr = qkv + (size_t)(b * SEQ + tok) * QKVLD + CH + h * HD;
#pragma unroll
      for (int ks = 0; ks < 2; ++ks)
        acc = wmma_bf16(qa[ks], load_b_frag(kptr + ks * 32, lane), acc);
      float pc = prior[b * SEQ + ct * 16 + m];
#pragma unroll
      for (int r = 0; r < 8; ++r) {
        float p = __expf(acc[r] * SCALE + pr[r] * pc - pm[r]) * pl[r];
        pb[w][r + ((lane >> 4) << 3)][t * 16 + m] = (__bf16)p;
      }
    }
    // re-layout P: C-layout -> A-layout (vector LDS reads, DS ops in-order per wave)
    const __bf16* prow = &pb[w][m][0] + ((lane >> 4) << 4);
    v16bf pa = cat8(*(const v8bf*)(prow), *(const v8bf*)(prow + 8));
#pragma unroll
    for (int dt = 0; dt < 4; ++dt) {               // d = 64 -> 4 col tiles
      const __bf16* vptr = vT + ((size_t)((b * NH + h) * HD + dt * 16 + m)) * SEQ +
                           c2 * 32;
      oacc[dt] = wmma_bf16(pa, load_b_frag(vptr, lane), oacc[dt]);
    }
  }
#pragma unroll
  for (int dt = 0; dt < 4; ++dt)
#pragma unroll
    for (int r = 0; r < 8; ++r) {
      int row = rt * 16 + r + ((lane >> 4) << 3);
      houtb[(size_t)(b * SEQ + row) * CH + h * HD + dt * 16 + m] = (__bf16)oacc[dt][r];
    }
}

// ---------- K6: pass-3 head-mean attention weights -> d_out tail ----------
__global__ void k_weights(const __bf16* __restrict__ qkv, const float* __restrict__ prior,
                          const float* __restrict__ rm, const float* __restrict__ rl,
                          float* __restrict__ wout) {
  int lane = threadIdx.x & 31;
  int wid = blockIdx.x * 8 + (threadIdx.x >> 5);   // [0, 512)
  int b = wid >> 8, rt = wid & 255;
  int m = lane & 15;
  v16bf qa[NH][2];
#pragma unroll
  for (int h = 0; h < NH; ++h)
#pragma unroll
    for (int ks = 0; ks < 2; ++ks)
      qa[h][ks] = load_a_frag(qkv + (size_t)(b * SEQ + rt * 16 + m) * QKVLD +
                              h * HD + ks * 32, lane);
  float pr[8], pm[NH][8], pl[NH][8];
#pragma unroll
  for (int r = 0; r < 8; ++r) {
    int row = rt * 16 + r + ((lane >> 4) << 3);
    pr[r] = prior[b * SEQ + row] * PSTR;
#pragma unroll
    for (int h = 0; h < NH; ++h) {
      size_t o = (size_t)(b * NH + h) * SEQ + row;
      pm[h][r] = rm[o];
      pl[h][r] = 1.f / rl[o];
    }
  }
  for (int ct = 0; ct < 256; ++ct) {
    float pc = prior[b * SEQ + ct * 16 + m];
    float wsum[8];
#pragma unroll
    for (int r = 0; r < 8; ++r) wsum[r] = 0.f;
    int tok = ct * 16 + m;
#pragma unroll
    for (int h = 0; h < NH; ++h) {
      v8f acc = {};
      const __bf16* kptr = qkv + (size_t)(b * SEQ + tok) * QKVLD + CH + h * HD;
#pragma unroll
      for (int ks = 0; ks < 2; ++ks)
        acc = wmma_bf16(qa[h][ks], load_b_frag(kptr + ks * 32, lane), acc);
#pragma unroll
      for (int r = 0; r < 8; ++r)
        wsum[r] += __expf(acc[r] * SCALE + pr[r] * pc - pm[h][r]) * pl[h][r];
    }
#pragma unroll
    for (int r = 0; r < 8; ++r) {
      int row = rt * 16 + r + ((lane >> 4) << 3);
      wout[(size_t)(b * SEQ + row) * SEQ + ct * 16 + m] = wsum[r] * 0.25f;
    }
  }
}

// ---------- K7: out = head_out @ Wproj + bproj -> d_out head ----------
__global__ void k_proj(const __bf16* __restrict__ houtb, const __bf16* __restrict__ WT,
                       const float* __restrict__ bias, float* __restrict__ out) {
  int lane = threadIdx.x & 31;
  int wid = blockIdx.x * 8 + (threadIdx.x >> 5);   // [0, 8192)
  int nt = wid & 15, rt = wid >> 4;                // 16 col tiles, 512 row tiles
  int m = lane & 15;
  int col = nt * 16 + m;
  v8f acc = {};
  for (int kt = 0; kt < 8; ++kt) {
    v16bf a = load_a_frag(houtb + (size_t)(rt * 16 + m) * CH + kt * 32, lane);
    v16bf b = load_b_frag(WT + (size_t)col * CH + kt * 32, lane);
    acc = wmma_bf16(a, b, acc);
  }
  float bv = bias[col];
#pragma unroll
  for (int r = 0; r < 8; ++r)
    out[(size_t)(rt * 16 + r + ((lane >> 4) << 3)) * CH + col] = acc[r] + bv;
}

extern "C" void kernel_launch(void* const* d_in, const int* in_sizes, int n_in,
                              void* d_out, int out_size, void* d_ws, size_t ws_size,
                              hipStream_t stream) {
  const float* x     = (const float*)d_in[0];
  const float* Wqkv  = (const float*)d_in[1];
  const float* bqkv  = (const float*)d_in[2];
  const float* Wproj = (const float*)d_in[3];
  const float* bproj = (const float*)d_in[4];
  const float* ak    = (const float*)d_in[5];
  const float* ck    = (const float*)d_in[6];
  (void)in_sizes; (void)n_in; (void)out_size; (void)ws_size;

  // workspace layout (bytes, all 32B-aligned)
  char* ws = (char*)d_ws;
  float*  xm     = (float*)(ws);                             // 32 KB
  float*  prior  = (float*)(ws + 32768);                     // 32 KB
  float*  rmax   = (float*)(ws + 65536);                     // 128 KB
  float*  rsum   = (float*)(ws + 196608);                    // 128 KB
  __bf16* qkv    = (__bf16*)(ws + 327680);                   // 12.6 MB
  __bf16* xb     = (__bf16*)(ws + 12910592);                 // 4 MB
  __bf16* WqkvT  = (__bf16*)(ws + 17104896);                 // 384 KB
  __bf16* WprojT = (__bf16*)(ws + 17498112);                 // 128 KB
  __bf16* vT     = (__bf16*)(ws + 17629184);                 // 4 MB
  __bf16* houtb  = (__bf16*)(ws + 21823488);                 // 4 MB

  float* out  = (float*)d_out;                               // [B,N,C]
  float* wout = out + (size_t)BATCH * SEQ * CH;              // [B,N,N]

  k_mean   <<<BATCH * SEQ / 256,           256, 0, stream>>>(x, xm);
  k_prior  <<<BATCH * SEQ / 256,           256, 0, stream>>>(xm, ak, ck, prior);
  k_xcvt   <<<BATCH * SEQ * CH / 4 / 256,  256, 0, stream>>>(x, xb);
  k_wt     <<<QKVLD * CH / 256,            256, 0, stream>>>(Wqkv, WqkvT, QKVLD);
  k_wt     <<<CH * CH / 256,               256, 0, stream>>>(Wproj, WprojT, CH);
  k_qkv    <<<512 * 48 / 8,                256, 0, stream>>>(xb, WqkvT, bqkv, qkv);
  k_vt     <<<BATCH * NH * HD * SEQ / 256, 256, 0, stream>>>(qkv, vT);
  k_stats  <<<2048 / 8,                    256, 0, stream>>>(qkv, prior, rmax, rsum);
  k_av     <<<2048 / 8,                    256, 0, stream>>>(qkv, vT, prior, rmax, rsum, houtb);
  k_weights<<<512 / 8,                     256, 0, stream>>>(qkv, prior, rmax, rsum, wout);
  k_proj   <<<8192 / 8,                    256, 0, stream>>>(houtb, WprojT, bproj, out);
}
